// AttentionLayer_12137577578670
// MI455X (gfx1250) — matrix operand order repaired and checked by reading, not compiled, not run on hardware
//
#include <hip/hip_runtime.h>
#include <hip/hip_bf16.h>
#include <math.h>

#define H_  8
#define HD  64
#define E_  512
#define L_  4096
#define B_  2
#define M_  (B_ * L_)   // 8192 rows for the projection GEMMs
#define LDP 72          // padded LDS row stride (halfs) -> conflict-free ds_load_b128

typedef __attribute__((ext_vector_type(16))) _Float16 v16h;
typedef __attribute__((ext_vector_type(8)))  _Float16 v8h;
typedef __attribute__((ext_vector_type(8)))  float    v8f;
typedef __attribute__((ext_vector_type(4)))  float    v4f;

static __device__ __forceinline__ v8f wmma_f16(v16h a, v16h b, v8f c) {
  // D = A(16x32 f16) * B(32x16 f16) + C(16x16 f32)
  return __builtin_amdgcn_wmma_f32_16x16x32_f16(false, a, false, b, (short)0, c,
                                                false, false);
}

// ---- A-fragment (16x32, MxK) loader ----------------------------------------
// Contiguous-k layout used here (A and B match): lanes 0-15 row M=lane, K=k0..k0+15;
//             lanes 16-31 row M=lane-16, K=k0+16..k0+31.
static __device__ __forceinline__ v16h load_a_f16(const _Float16* base,
                                                  int ld, int r0, int k0) {
  int lane = threadIdx.x & 31;
  const _Float16* p = base + (r0 + (lane & 15)) * ld + k0 + ((lane & 16) ? 16 : 0);
  v8h lo = *(const v8h*)p;          // K = c .. c+7
  v8h hi = *(const v8h*)(p + 8);    // K = c+8 .. c+15
  v16h o;
#pragma unroll
  for (int i = 0; i < 8; ++i) { o[i] = lo[i]; o[i + 8] = hi[i]; }
  return o;
}

// ---- B-fragment (32x16, KxN) loader, where B[k][n] = src[n][k] -------------
// Contiguous-k layout used here (matches A): lanes 0-15 col N=lane, K=k0..k0+15; lanes 16-31 col N=lane-16,
// K=k0+16..k0+31 -> per lane 16 consecutive K values (contiguous in src rows).
static __device__ __forceinline__ v16h load_b_f16rows(const _Float16* base,
                                                      int ld, int n0, int k0) {
  int lane = threadIdx.x & 31;
  const _Float16* p = base + (n0 + (lane & 15)) * ld + k0 + ((lane & 16) ? 16 : 0);
  v8h lo = *(const v8h*)p;
  v8h hi = *(const v8h*)(p + 8);
  v16h o;
#pragma unroll
  for (int i = 0; i < 8; ++i) { o[i] = lo[i]; o[i + 8] = hi[i]; }
  return o;
}

// Convert two 16x16 S^T C-fragments (f32 probabilities; tile t covers keys
// t*16+{0..7} in lanes 0-15 and t*16+{8..15} in lanes 16-31) into one P^T
// B-fragment covering 32 consecutive keys, using a single xor-16 exchange.
static __device__ __forceinline__ v16h build_p_bfrag(v8f p0, v8f p1) {
  bool lo = (threadIdx.x & 16) == 0;
  v16h o;
#pragma unroll
  for (int i = 0; i < 8; ++i) {
    float send = lo ? p1[i] : p0[i];
    float recv = __shfl_xor(send, 16, 32);
    float own  = lo ? p0[i] : p1[i];
    o[i]     = (_Float16)(lo ? own  : recv);
    o[i + 8] = (_Float16)(lo ? recv : own);
  }
  return o;
}

// ======================= Kernel 0: f32 -> f16 conversion ====================
__global__ __launch_bounds__(256)
void cvt_kernel(const float* __restrict__ src, _Float16* __restrict__ dst, int n) {
  int i = (blockIdx.x * 256 + threadIdx.x) * 8;
  if (i >= n) return;
  v4f a = *(const v4f*)(src + i);
  v4f b = *(const v4f*)(src + i + 4);
  v8h o;
#pragma unroll
  for (int j = 0; j < 4; ++j) { o[j] = (_Float16)a[j]; o[j + 4] = (_Float16)b[j]; }
  *(v8h*)(dst + i) = o;
}

// ======================= Kernel 1: QKV projections ==========================
// out[m,n] = sum_k X[m,k] * W[n,k] + b[n]  (torch Linear), f16 in/out of WMMA.
// Wave tile 32m x 64n (8 WMMA per K-step). Mode 2 (V) writes Vt[b][h][d][s].
__global__ __launch_bounds__(128)
void qkv_proj_kernel(const _Float16* __restrict__ Xq, const _Float16* __restrict__ Xk,
                     const _Float16* __restrict__ Xv,
                     const _Float16* __restrict__ Wq, const _Float16* __restrict__ Wk,
                     const _Float16* __restrict__ Wv,
                     const float* __restrict__ bq, const float* __restrict__ bk,
                     const float* __restrict__ bv,
                     _Float16* __restrict__ Qh, _Float16* __restrict__ Kh,
                     _Float16* __restrict__ Vt) {
  int mode = blockIdx.z;
  const _Float16* X    = mode == 0 ? Xq : mode == 1 ? Xk : Xv;
  const _Float16* W    = mode == 0 ? Wq : mode == 1 ? Wk : Wv;
  const float*    bias = mode == 0 ? bq : mode == 1 ? bk : bv;

  int wave = threadIdx.x >> 5;
  int lane = threadIdx.x & 31;
  int m0 = blockIdx.x * 128 + wave * 32;  // row tile (over B*L)
  int n0 = blockIdx.y * 64;               // col tile (over E)

  v8f c[2][4];
#pragma unroll
  for (int mi = 0; mi < 2; ++mi)
#pragma unroll
    for (int t = 0; t < 4; ++t) c[mi][t] = v8f{};

  for (int k0 = 0; k0 < E_; k0 += 32) {
    v16h a0 = load_a_f16(X, E_, m0, k0);
    v16h a1 = load_a_f16(X, E_, m0 + 16, k0);
#pragma unroll
    for (int t = 0; t < 4; ++t) {
      v16h bf = load_b_f16rows(W, E_, n0 + t * 16, k0);
      c[0][t] = wmma_f16(a0, bf, c[0][t]);
      c[1][t] = wmma_f16(a1, bf, c[1][t]);
    }
  }

  int nlane = lane & 15;
  int roff  = (lane & 16) ? 8 : 0;
#pragma unroll
  for (int mi = 0; mi < 2; ++mi) {
    int mb = m0 + mi * 16;
    if (mode < 2) {
      _Float16* dst = mode == 0 ? Qh : Kh;
#pragma unroll
      for (int t = 0; t < 4; ++t) {
        int n = n0 + t * 16 + nlane;
        int h = n >> 6, d = n & 63;
        float bn = bias[n];
#pragma unroll
        for (int vv = 0; vv < 8; ++vv) {
          int m = mb + roff + vv;
          int bb = m >> 12, l = m & (L_ - 1);
          dst[(((bb * H_ + h) * L_ + l) << 6) + d] = (_Float16)(c[mi][t][vv] + bn);
        }
      }
    } else {
      // Vt[b][h][d][s]: 8 consecutive s per lane -> one 16B store
#pragma unroll
      for (int t = 0; t < 4; ++t) {
        int n = n0 + t * 16 + nlane;
        int h = n >> 6, d = n & 63;
        float bn = bias[n];
        int m = mb + roff;
        int bb = m >> 12, s = m & (L_ - 1);
        v8h o;
#pragma unroll
        for (int vv = 0; vv < 8; ++vv) o[vv] = (_Float16)(c[mi][t][vv] + bn);
        *(v8h*)(Vt + (size_t)((bb * H_ + h) * HD + d) * L_ + s) = o;
      }
    }
  }
}

// ======================= Kernel 2: fused flash attention ====================
// Block = 8 waves = 128 queries of one (b,h); K/V 64-key tiles staged in LDS
// (rows padded to LDP=72 halfs -> conflict-free b128 LDS reads).
// Per wave: S^T = K*Q^T (lane owns a query), per-lane online softmax
// (+1 shfl_xor(16) to merge the lane-pair halves), O^T = V^T * P^T in regs.
__global__ __launch_bounds__(256)
void attn_kernel(const _Float16* __restrict__ Qh, const _Float16* __restrict__ Kh,
                 const _Float16* __restrict__ Vt, _Float16* __restrict__ AO) {
  __shared__ _Float16 tK[64 * LDP];  // [key_local][d]
  __shared__ _Float16 tV[64 * LDP];  // [d][key_local]

  int b = blockIdx.z, h = blockIdx.y;
  int tid  = threadIdx.x;
  int wave = tid >> 5;
  int lane = tid & 31;
  int q0 = (blockIdx.x * 8 + wave) * 16;

  const _Float16* Qbh = Qh + (size_t)((b * H_ + h) * L_) * HD;
  const _Float16* Kbh = Kh + (size_t)((b * H_ + h) * L_) * HD;
  const _Float16* Vbh = Vt + (size_t)((b * H_ + h) * HD) * L_;

  // Q B-fragments (contraction over d=64 -> two K-steps), kept for whole loop
  v16h bq0 = load_b_f16rows(Qbh, HD, q0, 0);
  v16h bq1 = load_b_f16rows(Qbh, HD, q0, 32);

  v8f acc[4];
#pragma unroll
  for (int t = 0; t < 4; ++t) acc[t] = v8f{};
  float mrun = -1e30f, lrun = 0.f;
  const float scale = 0.125f;  // hd^-0.5

  for (int kb = 0; kb < L_; kb += 64) {
    __syncthreads();
    // cooperative stage: 4096 halfs per tile, 16 halfs per thread
#pragma unroll
    for (int j = 0; j < 2; ++j) {
      int e = tid * 16 + j * 8;
      int r = e >> 6, cc = e & 63;
      *(v8h*)(tK + r * LDP + cc) = *(const v8h*)(Kbh + (size_t)(kb + r) * HD + cc);
      *(v8h*)(tV + r * LDP + cc) = *(const v8h*)(Vbh + (size_t)r * L_ + kb + cc);
    }
    __syncthreads();

    // S^T tiles: 64 keys x 16 queries
    v8f ct[4];
#pragma unroll
    for (int t = 0; t < 4; ++t) {
      v8f z = v8f{};
      v16h a0 = load_a_f16(tK, LDP, t * 16, 0);
      z = wmma_f16(a0, bq0, z);
      v16h a1 = load_a_f16(tK, LDP, t * 16, 32);
      ct[t] = wmma_f16(a1, bq1, z);
    }

    // online softmax: per-lane over its 32 key-values, merged across lane pair
    float vmax = -1e30f;
#pragma unroll
    for (int t = 0; t < 4; ++t)
#pragma unroll
      for (int i = 0; i < 8; ++i) vmax = fmaxf(vmax, ct[t][i]);
    vmax *= scale;
    vmax = fmaxf(vmax, __shfl_xor(vmax, 16, 32));
    float mnew  = fmaxf(mrun, vmax);
    float alpha = __expf(mrun - mnew);
    mrun = mnew;
    lrun *= alpha;
#pragma unroll
    for (int t = 0; t < 4; ++t)
#pragma unroll
      for (int i = 0; i < 8; ++i) acc[t][i] *= alpha;

    float ps = 0.f;
#pragma unroll
    for (int t = 0; t < 4; ++t)
#pragma unroll
      for (int i = 0; i < 8; ++i) {
        float p = __expf(ct[t][i] * scale - mnew);
        ct[t][i] = p;
        ps += p;
      }
    lrun += ps + __shfl_xor(ps, 16, 32);

    // O^T += V^T * P^T  (two 32-key steps, 4 d-tiles each)
#pragma unroll
    for (int s = 0; s < 2; ++s) {
      v16h bp = build_p_bfrag(ct[2 * s], ct[2 * s + 1]);
#pragma unroll
      for (int t = 0; t < 4; ++t) {
        v16h av = load_a_f16(tV, LDP, t * 16, s * 32);
        acc[t] = wmma_f16(av, bp, acc[t]);
      }
    }
  }

  float inv = 1.f / lrun;
  int qg   = q0 + (lane & 15);
  int doff = (lane & 16) ? 8 : 0;
  _Float16* dst = AO + (size_t)(b * L_ + qg) * E_ + h * HD;
#pragma unroll
  for (int t = 0; t < 4; ++t) {
    v8h o;
#pragma unroll
    for (int i = 0; i < 8; ++i) o[i] = (_Float16)(acc[t][i] * inv);
    *(v8h*)(dst + t * 16 + doff) = o;
  }
}

// ======================= Kernel 3: output projection ========================
__global__ __launch_bounds__(128)
void out_proj_kernel(const _Float16* __restrict__ AO, const _Float16* __restrict__ Wo,
                     const float* __restrict__ bo, float* __restrict__ out) {
  int wave = threadIdx.x >> 5;
  int lane = threadIdx.x & 31;
  int m0 = blockIdx.x * 128 + wave * 32;
  int n0 = blockIdx.y * 64;

  v8f c[2][4];
#pragma unroll
  for (int mi = 0; mi < 2; ++mi)
#pragma unroll
    for (int t = 0; t < 4; ++t) c[mi][t] = v8f{};

  for (int k0 = 0; k0 < E_; k0 += 32) {
    v16h a0 = load_a_f16(AO, E_, m0, k0);
    v16h a1 = load_a_f16(AO, E_, m0 + 16, k0);
#pragma unroll
    for (int t = 0; t < 4; ++t) {
      v16h bf = load_b_f16rows(Wo, E_, n0 + t * 16, k0);
      c[0][t] = wmma_f16(a0, bf, c[0][t]);
      c[1][t] = wmma_f16(a1, bf, c[1][t]);
    }
  }

  int nlane = lane & 15;
  int roff  = (lane & 16) ? 8 : 0;
#pragma unroll
  for (int mi = 0; mi < 2; ++mi) {
#pragma unroll
    for (int t = 0; t < 4; ++t) {
      int n = n0 + t * 16 + nlane;
      float bn = bo[n];
#pragma unroll
      for (int vv = 0; vv < 8; ++vv)
        out[(size_t)(m0 + mi * 16 + roff + vv) * E_ + n] = c[mi][t][vv] + bn;
    }
  }
}

// ============================================================================
extern "C" void kernel_launch(void* const* d_in, const int* in_sizes, int n_in,
                              void* d_out, int out_size, void* d_ws, size_t ws_size,
                              hipStream_t stream) {
  const float* q  = (const float*)d_in[0];
  const float* k  = (const float*)d_in[1];
  const float* v  = (const float*)d_in[2];
  const float* Wq = (const float*)d_in[3];
  const float* bq = (const float*)d_in[4];
  const float* Wk = (const float*)d_in[5];
  const float* bk = (const float*)d_in[6];
  const float* Wv = (const float*)d_in[7];
  const float* bv = (const float*)d_in[8];
  const float* Wo = (const float*)d_in[9];
  const float* bo = (const float*)d_in[10];
  float* out = (float*)d_out;

  // Workspace (f16): 7 big arrays of M_*E_ = 4M halfs + 4 weight arrays of 256K
  size_t big = (size_t)M_ * E_;      // == B_*H_*L_*HD
  size_t wsz = (size_t)E_ * E_;
  _Float16* p    = (_Float16*)d_ws;
  _Float16* Qh   = p;        p += big;
  _Float16* Kh   = p;        p += big;
  _Float16* Vt   = p;        p += big;
  _Float16* AO   = p;        p += big;
  _Float16* Xq16 = p;        p += big;
  _Float16* Xk16 = p;        p += big;
  _Float16* Xv16 = p;        p += big;
  _Float16* Wq16 = p;        p += wsz;
  _Float16* Wk16 = p;        p += wsz;
  _Float16* Wv16 = p;        p += wsz;
  _Float16* Wo16 = p;        p += wsz;

  int nb_big = (int)(big / 8 / 256);   // 2048 blocks
  int nb_w   = (int)(wsz / 8 / 256);   // 128 blocks
  cvt_kernel<<<nb_big, 256, 0, stream>>>(q,  Xq16, (int)big);
  cvt_kernel<<<nb_big, 256, 0, stream>>>(k,  Xk16, (int)big);
  cvt_kernel<<<nb_big, 256, 0, stream>>>(v,  Xv16, (int)big);
  cvt_kernel<<<nb_w,   256, 0, stream>>>(Wq, Wq16, (int)wsz);
  cvt_kernel<<<nb_w,   256, 0, stream>>>(Wk, Wk16, (int)wsz);
  cvt_kernel<<<nb_w,   256, 0, stream>>>(Wv, Wv16, (int)wsz);
  cvt_kernel<<<nb_w,   256, 0, stream>>>(Wo, Wo16, (int)wsz);

  qkv_proj_kernel<<<dim3(M_ / 128, E_ / 64, 3), 128, 0, stream>>>(
      Xq16, Xk16, Xv16, Wq16, Wk16, Wv16, bq, bk, bv, Qh, Kh, Vt);
  attn_kernel<<<dim3(L_ / 128, H_, B_), 256, 0, stream>>>(Qh, Kh, Vt, AO);
  out_proj_kernel<<<dim3(M_ / 128, E_ / 64), 128, 0, stream>>>(AO, Wo16, bo, out);
}
